// EEGGraphModel_1640677507489
// MI455X (gfx1250) — matrix operand order repaired and checked by reading, not compiled, not run on hardware
//
#include <hip/hip_runtime.h>

typedef _Float16 h16;
typedef __attribute__((ext_vector_type(16))) _Float16 v16h;
typedef __attribute__((ext_vector_type(8)))  _Float16 v8h;
typedef __attribute__((ext_vector_type(4)))  _Float16 v4h;
typedef __attribute__((ext_vector_type(8)))  float    v8f;
typedef __attribute__((ext_vector_type(4)))  float    v4f;

#define T_IN    100000
#define L1      20000
#define L2      4000
#define L3      800
#define NCH     256
#define CNN_OUT 32
#define GH      64
#define G3      192

// ---- conv tiling ----
#define TILE3   50
#define NTILE   16
#define C2LEN   250
#define C1LEN   1252
#define RAWLEN  6266

// ---- workspace byte offsets (all 256B aligned) ----
#define WS_XS     0ULL            // h16 [800][256][32]
#define WS_WIH16  13107200ULL     // h16 [2][192][32]
#define WS_WHH16  13131776ULL     // h16 [2][192][64]
#define WS_HN     13180928ULL     // f32 [256][128]
#define WS_XN     13312000ULL     // h16 [256][128]
#define WS_CORR   13377536ULL     // f32 [256][256]
#define WS_DINV   13639680ULL     // f32 [256]
#define WS_TMP1   13640704ULL     // f32 [256][32]
#define WS_H1     13673472ULL     // f32 [256][32]
#define WS_TMP2   13706240ULL     // f32 [256][32]
#define WS_H2     13739008ULL     // f32 [256][32]

#define CONV_LDS_BYTES 95528
#define GRU_LDS_BYTES  70144

#if defined(__AMDGCN__) && __has_builtin(__builtin_amdgcn_global_load_async_to_lds_b32)
#define HAVE_ASYNC_LDS 1
typedef __attribute__((address_space(1))) int as1_int;
typedef __attribute__((address_space(3))) int as3_int;
#else
#define HAVE_ASYNC_LDS 0
#endif

// ================= fast math helpers =================
__device__ __forceinline__ float fast_sigmoid(float x) {
  float e = __expf(-x);
#if defined(__AMDGCN__) && __has_builtin(__builtin_amdgcn_rcpf)
  return __builtin_amdgcn_rcpf(1.0f + e);
#else
  return 1.0f / (1.0f + e);
#endif
}
__device__ __forceinline__ float fast_tanh(float x) {
#if defined(__AMDGCN__) && __has_builtin(__builtin_amdgcn_tanhf)
  return __builtin_amdgcn_tanhf(x);
#else
  return tanhf(x);
#endif
}

// ================= WMMA helpers =================
__device__ __forceinline__ v8f wmma32f16(v16h a, v16h b, v8f c) {
  return __builtin_amdgcn_wmma_f32_16x16x32_f16(false, a, false, b, (short)0, c,
                                                false, false);
}

__device__ __forceinline__ v16h mk16(const h16* p0, const h16* p1) {
  v8h a = *(const v8h*)p0;
  v8h b = *(const v8h*)p1;
  return __builtin_shufflevector(a, b, 0,1,2,3,4,5,6,7,8,9,10,11,12,13,14,15);
}

// A-operand (16xK tile, row-major src): lanes0-15 -> K {0-7,16-23}, lanes16-31 -> K {8-15,24-31}
__device__ __forceinline__ v16h load_A(const h16* base, int stride, int lane, int koff) {
  int row = lane & 15;
  int k0  = koff + ((lane < 16) ? 0 : 8);
  const h16* r = base + row * stride;
  return mk16(r + k0, r + k0 + 16);
}

// B-operand (Kx16 tile from W[n][k] row-major): lane holds col n, lanes0-15 -> K {0-7,16-23}, lanes16-31 -> K {8-15,24-31}
__device__ __forceinline__ v16h load_B(const h16* wbase, int stride, int lane,
                                       int ncol0, int koff) {
  int col = ncol0 + (lane & 15);
  int k0  = koff + ((lane < 16) ? 0 : 8);
  const h16* r = wbase + col * stride;
  return mk16(r + k0, r + k0 + 16);
}

// ================= fused conv1/conv2/conv3 =================
__global__ void conv_fused_kernel(const float* __restrict__ data,
                                  const float* __restrict__ c1w, const float* __restrict__ c1b,
                                  const float* __restrict__ c2w, const float* __restrict__ c2b,
                                  const float* __restrict__ c3w, const float* __restrict__ c3b,
                                  h16* __restrict__ xs) {
  extern __shared__ __align__(16) unsigned char smem[];
  float* raw = (float*)smem;          // 6266
  float* sc1 = raw + RAWLEN;          // 8*1252
  float* sc2 = sc1 + 8 * C1LEN;       // 16*250
  float* w1  = sc2 + 16 * C2LEN;      // 88
  float* b1  = w1 + 88;
  float* w2  = b1 + 8;                // 896
  float* b2  = w2 + 896;
  float* w3  = b2 + 16;               // 2560
  float* b3  = w3 + 2560;

  const int tid = threadIdx.x;
  const int ch  = blockIdx.x;
  const float* dch = data + (size_t)ch * T_IN;

  for (int i = tid; i < 88;   i += 256) w1[i] = c1w[i];
  for (int i = tid; i < 8;    i += 256) b1[i] = c1b[i];
  for (int i = tid; i < 896;  i += 256) w2[i] = c2w[i];
  for (int i = tid; i < 16;   i += 256) b2[i] = c2b[i];
  for (int i = tid; i < 2560; i += 256) w3[i] = c3w[i];
  for (int i = tid; i < 32;   i += 256) b3[i] = c3b[i];
  __syncthreads();

  for (int tile = 0; tile < NTILE; ++tile) {
    const int t3s = tile * TILE3;
    const int s2  = 5 * t3s - 2;
    const int s1  = 5 * s2 - 3;
    const int s0  = 5 * s1 - 5;

    // stage raw samples (zero padded) via async global->LDS when available
#if HAVE_ASYNC_LDS
    for (int j = tid; j < RAWLEN; j += 256) {
      int g = s0 + j;
      if (g >= 0 && g < T_IN) {
        __builtin_amdgcn_global_load_async_to_lds_b32(
            (as1_int*)(unsigned long long)(dch + g),
            (as3_int*)(raw + j), 0, 0);
      } else {
        raw[j] = 0.0f;
      }
    }
#if __has_builtin(__builtin_amdgcn_s_wait_asynccnt)
    __builtin_amdgcn_s_wait_asynccnt(0);
#else
    asm volatile("s_wait_asynccnt 0x0" ::: "memory");
#endif
#else
    for (int j = tid; j < RAWLEN; j += 256) {
      int g = s0 + j;
      raw[j] = (g >= 0 && g < T_IN) ? dch[g] : 0.0f;
      __builtin_prefetch(dch + g + 6250, 0, 0);
    }
#endif
    __syncthreads();

    // conv1: 8 x 1252
    for (int idx = tid; idx < 8 * C1LEN; idx += 256) {
      int oc = idx / C1LEN, j = idx - oc * C1LEN;
      int i1 = s1 + j;
      float v = 0.0f;
      if (i1 >= 0 && i1 < L1) {
        float s = b1[oc];
        const float* wr = w1 + oc * 11;
        const float* rp = raw + 5 * j;
#pragma unroll
        for (int k = 0; k < 11; ++k) s += rp[k] * wr[k];
        v = fmaxf(s, 0.0f);
      }
      sc1[idx] = v;
    }
    __syncthreads();

    // conv2: 16 x 250
    for (int idx = tid; idx < 16 * C2LEN; idx += 256) {
      int oc = idx / C2LEN, j = idx - oc * C2LEN;
      int i2 = s2 + j;
      float v = 0.0f;
      if (i2 >= 0 && i2 < L2) {
        float s = b2[oc];
#pragma unroll
        for (int ic = 0; ic < 8; ++ic) {
          const float* rp = sc1 + ic * C1LEN + 5 * j;
          const float* wr = w2 + (oc * 8 + ic) * 7;
#pragma unroll
          for (int k = 0; k < 7; ++k) s += rp[k] * wr[k];
        }
        v = fmaxf(s, 0.0f);
      }
      sc2[idx] = v;
    }
    __syncthreads();

    // conv3: 32 x 50 -> f16 xs[t][ch][oc]
    for (int idx = tid; idx < 32 * TILE3; idx += 256) {
      int oc = idx / TILE3, j = idx - oc * TILE3;
      float s = b3[oc];
#pragma unroll
      for (int ic = 0; ic < 16; ++ic) {
        const float* rp = sc2 + ic * C2LEN + 5 * j;
        const float* wr = w3 + (oc * 16 + ic) * 5;
#pragma unroll
        for (int k = 0; k < 5; ++k) s += rp[k] * wr[k];
      }
      s = fmaxf(s, 0.0f);
      xs[((size_t)(t3s + j) * NCH + ch) * CNN_OUT + oc] = (h16)s;
    }
    __syncthreads();
  }
}

// ================= GRU weight f32->f16 prep =================
__global__ void prep_gru_weights(const float* __restrict__ wih_f, const float* __restrict__ whh_f,
                                 const float* __restrict__ wih_b, const float* __restrict__ whh_b,
                                 h16* __restrict__ Wih16, h16* __restrict__ Whh16) {
  int idx = blockIdx.x * blockDim.x + threadIdx.x;
  if (idx < 6144)        Wih16[idx]          = (h16)wih_f[idx];
  else if (idx < 12288)  Wih16[idx]          = (h16)wih_b[idx - 6144];
  else if (idx < 24576)  Whh16[idx - 12288]  = (h16)whh_f[idx - 12288];
  else if (idx < 36864)  Whh16[idx - 24576]  = (h16)whh_b[idx - 24576];
}

// ================= GRU scan (WMMA, software-pipelined x loads) =================
// 32 blocks x 384 threads (12 waves, 1 N-tile each); 800 sequential steps
__global__ void gru_scan_kernel(const h16* __restrict__ xs,
                                const h16* __restrict__ Wih16, const h16* __restrict__ Whh16,
                                const float* __restrict__ bih_f, const float* __restrict__ bhh_f,
                                const float* __restrict__ bih_b, const float* __restrict__ bhh_b,
                                float* __restrict__ hn) {
  extern __shared__ __align__(16) unsigned char smem[];
  h16*   sWih = (h16*)smem;           // 192*32
  h16*   sWhh = sWih + 6144;          // 192*64
  h16*   sAx  = sWhh + 12288;         // 16*32
  h16*   sAh  = sAx + 512;            // 16*64
  float* sGi  = (float*)(sAh + 1024); // 16*192
  float* sGh  = sGi + 3072;           // 16*192
  float* sH   = sGh + 3072;           // 16*64
  float* sBi  = sH + 1024;            // 192
  float* sBh  = sBi + 192;            // 192

  const int tid  = threadIdx.x;
  const int lane = tid & 31;
  const int wave = tid >> 5;          // 0..11, one 16-col tile each
  const int dir  = blockIdx.x >> 4;
  const int ch0  = (blockIdx.x & 15) * 16;

  const h16*   wih = Wih16 + dir * 6144;
  const h16*   whh = Whh16 + dir * 12288;
  const float* bi  = dir ? bih_b : bih_f;
  const float* bh  = dir ? bhh_b : bhh_f;

  for (int i = tid; i < 6144;  i += 384) sWih[i] = wih[i];
  for (int i = tid; i < 12288; i += 384) sWhh[i] = whh[i];
  for (int i = tid; i < 192;   i += 384) { sBi[i] = bi[i]; sBh[i] = bh[i]; }
  for (int i = tid; i < 1024;  i += 384) { sH[i] = 0.0f; sAh[i] = (h16)0.0f; }
  __syncthreads();

  // register-pipelined x_t fetch: threads 0..127 each carry 4 halves
  const int id4  = (tid & 127) * 4;
  const int xrow = id4 >> 5;
  const int xcol = id4 & 31;
  v4h xreg = {};
  if (tid < 128) {
    int t0 = dir ? (L3 - 1) : 0;
    xreg = *(const v4h*)(xs + ((size_t)t0 * NCH + ch0 + xrow) * CNN_OUT + xcol);
  }

  for (int step = 0; step < L3; ++step) {
    if (tid < 128) *(v4h*)(sAx + xrow * 32 + xcol) = xreg;
    __syncthreads();

    // issue next step's load early; hides HBM latency behind GEMM + gates
    if (step + 1 < L3 && tid < 128) {
      int tn = dir ? (L3 - 2 - step) : (step + 1);
      xreg = *(const v4h*)(xs + ((size_t)tn * NCH + ch0 + xrow) * CNN_OUT + xcol);
    }

    // GEMM: Gi = x @ Wih^T (K=32), Gh = h @ Whh^T (K=64); one N-tile per wave
    {
      const int nt = wave;
      v8f gi = {};
      v8f gh = {};
      v16h ax = load_A(sAx, 32, lane, 0);
      v16h bx = load_B(sWih, 32, lane, nt * 16, 0);
      gi = wmma32f16(ax, bx, gi);
      v16h ah0 = load_A(sAh, 64, lane, 0);
      v16h bh0 = load_B(sWhh, 64, lane, nt * 16, 0);
      gh = wmma32f16(ah0, bh0, gh);
      v16h ah1 = load_A(sAh, 64, lane, 32);
      v16h bh1 = load_B(sWhh, 64, lane, nt * 16, 32);
      gh = wmma32f16(ah1, bh1, gh);

      const int N     = nt * 16 + (lane & 15);
      const int mbase = (lane < 16) ? 0 : 8;
#pragma unroll
      for (int i = 0; i < 8; ++i) {
        sGi[(mbase + i) * G3 + N] = gi[i];
        sGh[(mbase + i) * G3 + N] = gh[i];
      }
    }
    __syncthreads();

    // gate nonlinearities + state update: threads 0..255 own 4 contiguous u each
    // -> all Gi/Gh/bias/h accesses become aligned v4f (ds_load_b128), fully unrolled
    if (tid < 256) {
      const int m  = tid >> 4;          // 0..15
      const int u0 = (tid & 15) * 4;    // 0,4,...,60
      const float* gi = sGi + m * G3;
      const float* gh = sGh + m * G3;
      v4f gir  = *(const v4f*)(gi + u0);
      v4f giz  = *(const v4f*)(gi + 64 + u0);
      v4f gin  = *(const v4f*)(gi + 128 + u0);
      v4f ghr  = *(const v4f*)(gh + u0);
      v4f ghz  = *(const v4f*)(gh + 64 + u0);
      v4f ghn  = *(const v4f*)(gh + 128 + u0);
      v4f bir  = *(const v4f*)(sBi + u0);
      v4f biz  = *(const v4f*)(sBi + 64 + u0);
      v4f bin_ = *(const v4f*)(sBi + 128 + u0);
      v4f bhr  = *(const v4f*)(sBh + u0);
      v4f bhz  = *(const v4f*)(sBh + 64 + u0);
      v4f bhn  = *(const v4f*)(sBh + 128 + u0);
      v4f hp   = *(const v4f*)(sH + m * GH + u0);
      v4f hv;
#pragma unroll
      for (int q = 0; q < 4; ++q) {
        float r  = fast_sigmoid(gir[q] + bir[q] + ghr[q] + bhr[q]);
        float z  = fast_sigmoid(giz[q] + biz[q] + ghz[q] + bhz[q]);
        float nn = fast_tanh(gin[q] + bin_[q] + r * (ghn[q] + bhn[q]));
        hv[q] = (1.0f - z) * nn + z * hp[q];
      }
      *(v4f*)(sH + m * GH + u0) = hv;
      v4h hh = { (h16)hv[0], (h16)hv[1], (h16)hv[2], (h16)hv[3] };
      *(v4h*)(sAh + m * GH + u0) = hh;
    }
    // no barrier needed here: next step's sAx store touches a disjoint buffer,
    // and the next GEMM is fenced by the barrier after that store
  }

  __syncthreads();
  for (int idx = tid; idx < 1024; idx += 384) {
    int m = idx >> 6, u = idx & 63;
    hn[(size_t)(ch0 + m) * 128 + dir * 64 + u] = sH[idx];
  }
}

// ================= row normalize (mean/std ddof=1) -> f16 =================
__global__ void rowstats_kernel(const float* __restrict__ hn, h16* __restrict__ xn) {
  __shared__ float red[128];
  const int i = blockIdx.x, tid = threadIdx.x;
  float v = hn[(size_t)i * 128 + tid];
  red[tid] = v; __syncthreads();
  for (int s = 64; s > 0; s >>= 1) { if (tid < s) red[tid] += red[tid + s]; __syncthreads(); }
  float mean = red[0] / 128.0f; __syncthreads();
  float c = v - mean;
  red[tid] = c * c; __syncthreads();
  for (int s = 64; s > 0; s >>= 1) { if (tid < s) red[tid] += red[tid + s]; __syncthreads(); }
  float stdv = sqrtf(red[0] / 127.0f);
  float inv  = 1.0f / (stdv + 1e-6f);
  xn[(size_t)i * 128 + tid] = (h16)(c * inv);
}

// ================= corr = Xn @ Xn^T / 128 (WMMA) =================
__global__ void corr_wmma_kernel(const h16* __restrict__ xn, float* __restrict__ corr) {
  const int tid  = threadIdx.x;
  const int lane = tid & 31;
  const int wave = tid >> 5;
  const int tileid = blockIdx.x * 4 + wave;   // 0..255
  const int ti = tileid >> 4, tj = tileid & 15;

  const h16* Arow = xn + (size_t)ti * 16 * 128;
  const h16* Brow = xn + (size_t)tj * 16 * 128;
  v8f acc = {};
#pragma unroll
  for (int kb = 0; kb < 4; ++kb) {
    v16h a = load_A(Arow, 128, lane, kb * 32);
    v16h b = load_B(Brow, 128, lane, 0, kb * 32);
    acc = wmma32f16(a, b, acc);
  }
  const int N     = tj * 16 + (lane & 15);
  const int mbase = ti * 16 + ((lane < 16) ? 0 : 8);
#pragma unroll
  for (int i = 0; i < 8; ++i)
    corr[(size_t)(mbase + i) * 256 + N] = acc[i] * (1.0f / 128.0f);
}

// ================= degree / dinv =================
__global__ void deg_kernel(const float* __restrict__ corr, float* __restrict__ dinv) {
  __shared__ float red[256];
  const int i = blockIdx.x, tid = threadIdx.x;
  red[tid] = (fabsf(corr[(size_t)i * 256 + tid]) > 0.7f) ? 1.0f : 0.0f;
  __syncthreads();
  for (int s = 128; s > 0; s >>= 1) { if (tid < s) red[tid] += red[tid + s]; __syncthreads(); }
  if (tid == 0) dinv[i] = rsqrtf(fmaxf(red[0], 1e-12f));
}

// ================= tmp = in @ W^T  (in: 256xK, W: 32xK) =================
__global__ void xw_kernel(const float* __restrict__ in, const float* __restrict__ w,
                          float* __restrict__ out, int K) {
  __shared__ float row[128];
  const int i = blockIdx.x, o = threadIdx.x;
  for (int k = o; k < K; k += 32) row[k] = in[(size_t)i * K + k];
  __syncthreads();
  float s = 0.0f;
  for (int k = 0; k < K; ++k) s += row[k] * w[o * K + k];
  out[(size_t)i * 32 + o] = s;
}

// ================= out = relu(A_hat @ tmp + b) =================
__global__ void agg_kernel(const float* __restrict__ corr, const float* __restrict__ dinv,
                           const float* __restrict__ tmp, const float* __restrict__ bias,
                           float* __restrict__ out) {
  __shared__ float arow[256];
  const int i = blockIdx.x, o = threadIdx.x;
  for (int j = o; j < 256; j += 32)
    arow[j] = (fabsf(corr[(size_t)i * 256 + j]) > 0.7f) ? dinv[j] : 0.0f;
  __syncthreads();
  const float di = dinv[i];
  float s = 0.0f;
  for (int j = 0; j < 256; ++j) s += arow[j] * tmp[(size_t)j * 32 + o];
  out[(size_t)i * 32 + o] = fmaxf(di * s + bias[o], 0.0f);
}

// ================= mean-pool + classifier =================
__global__ void finale_kernel(const float* __restrict__ h2, const float* __restrict__ cls_w,
                              const float* __restrict__ cls_b, float* __restrict__ out) {
  __shared__ float e[32];
  const int o = threadIdx.x;
  float s = 0.0f;
  for (int i = 0; i < 256; ++i) s += h2[(size_t)i * 32 + o];
  e[o] = s * (1.0f / 256.0f);
  __syncthreads();
  if (o < 2) {
    float r = cls_b[o];
    for (int k = 0; k < 32; ++k) r += e[k] * cls_w[o * 32 + k];
    out[o] = r;
  }
}

extern "C" void kernel_launch(void* const* d_in, const int* in_sizes, int n_in,
                              void* d_out, int out_size, void* d_ws, size_t ws_size,
                              hipStream_t stream) {
  const float* data    = (const float*)d_in[0];
  const float* c1w     = (const float*)d_in[1];
  const float* c1b     = (const float*)d_in[2];
  const float* c2w     = (const float*)d_in[3];
  const float* c2b     = (const float*)d_in[4];
  const float* c3w     = (const float*)d_in[5];
  const float* c3b     = (const float*)d_in[6];
  const float* wih_f   = (const float*)d_in[7];
  const float* whh_f   = (const float*)d_in[8];
  const float* bih_f   = (const float*)d_in[9];
  const float* bhh_f   = (const float*)d_in[10];
  const float* wih_b   = (const float*)d_in[11];
  const float* whh_b   = (const float*)d_in[12];
  const float* bih_b   = (const float*)d_in[13];
  const float* bhh_b   = (const float*)d_in[14];
  const float* gcn1_w  = (const float*)d_in[15];
  const float* gcn1_b  = (const float*)d_in[16];
  const float* gcn2_w  = (const float*)d_in[17];
  const float* gcn2_b  = (const float*)d_in[18];
  const float* cls_w   = (const float*)d_in[19];
  const float* cls_b   = (const float*)d_in[20];

  unsigned char* ws = (unsigned char*)d_ws;
  h16*   xs    = (h16*)(ws + WS_XS);
  h16*   Wih16 = (h16*)(ws + WS_WIH16);
  h16*   Whh16 = (h16*)(ws + WS_WHH16);
  float* hn    = (float*)(ws + WS_HN);
  h16*   xn    = (h16*)(ws + WS_XN);
  float* corr  = (float*)(ws + WS_CORR);
  float* dinv  = (float*)(ws + WS_DINV);
  float* tmp1  = (float*)(ws + WS_TMP1);
  float* h1    = (float*)(ws + WS_H1);
  float* tmp2  = (float*)(ws + WS_TMP2);
  float* h2    = (float*)(ws + WS_H2);
  float* out   = (float*)d_out;

  prep_gru_weights<<<144, 256, 0, stream>>>(wih_f, whh_f, wih_b, whh_b, Wih16, Whh16);
  conv_fused_kernel<<<NCH, 256, CONV_LDS_BYTES, stream>>>(data, c1w, c1b, c2w, c2b, c3w, c3b, xs);
  gru_scan_kernel<<<32, 384, GRU_LDS_BYTES, stream>>>(xs, Wih16, Whh16,
                                                      bih_f, bhh_f, bih_b, bhh_b, hn);
  rowstats_kernel<<<NCH, 128, 0, stream>>>(hn, xn);
  corr_wmma_kernel<<<64, 128, 0, stream>>>(xn, corr);
  deg_kernel<<<NCH, 256, 0, stream>>>(corr, dinv);
  xw_kernel<<<NCH, 32, 0, stream>>>(hn, gcn1_w, tmp1, 128);
  agg_kernel<<<NCH, 32, 0, stream>>>(corr, dinv, tmp1, gcn1_b, h1);
  xw_kernel<<<NCH, 32, 0, stream>>>(h1, gcn2_w, tmp2, 32);
  agg_kernel<<<NCH, 32, 0, stream>>>(corr, dinv, tmp2, gcn2_b, h2);
  finale_kernel<<<1, 32, 0, stream>>>(h2, cls_w, cls_b, out);
}
